// Proposed_Model_51762945852058
// MI455X (gfx1250) — matrix operand order repaired and checked by reading, not compiled, not run on hardware
//
#include <hip/hip_runtime.h>
#include <hip/hip_bf16.h>
#include <math.h>

// ---------------------------------------------------------------------------
// Types for CDNA5 WMMA
// ---------------------------------------------------------------------------
typedef __attribute__((ext_vector_type(16))) __bf16 v16bf;
typedef __attribute__((ext_vector_type(8)))  float  v8f;
typedef unsigned short u16;

__device__ __forceinline__ u16 f2bf(float f) {
    unsigned x = __float_as_uint(f);
    unsigned r = (x + 0x7FFFu + ((x >> 16) & 1u)) >> 16;   // RNE
    return (u16)r;
}
__device__ __forceinline__ float bf2f(u16 h) {
    return __uint_as_float(((unsigned)h) << 16);
}

// block-wide sum reduce; blockDim.x == N (power of two); all threads call
template<int N>
__device__ __forceinline__ float block_reduce_sum(float v, float* red) {
    int t = threadIdx.x;
    red[t] = v; __syncthreads();
    for (int o = N >> 1; o > 0; o >>= 1) {
        if (t < o) red[t] += red[t + o];
        __syncthreads();
    }
    float r = red[0]; __syncthreads();
    return r;
}

// ---------------------------------------------------------------------------
// VGG config
// ---------------------------------------------------------------------------
static const int CFG_CI[16] = {3,64,64,128,128,256,256,256,256,512,512,512,512,512,512,512};
static const int CFG_CO[16] = {64,64,128,128,256,256,256,256,512,512,512,512,512,512,512,512};

// ---------------------------------------------------------------------------
// Zero-fill a u32 region (used for the zero page feeding padded WMMA A loads)
// ---------------------------------------------------------------------------
__global__ void zero_u32_kernel(unsigned* __restrict__ p, int n) {
    int i = blockIdx.x * 256 + threadIdx.x;
    if (i < n) p[i] = 0u;
}

// ---------------------------------------------------------------------------
// Layer 0: direct conv 3->64, NCHW f32 -> NHWC bf16, ReLU
// ---------------------------------------------------------------------------
__global__ void conv0_kernel(const float* __restrict__ x, const float* __restrict__ w,
                             const float* __restrict__ bias, u16* __restrict__ out, int Bn) {
    long i = (long)blockIdx.x * 256 + threadIdx.x;
    long total = (long)Bn * 224 * 224 * 64;
    if (i >= total) return;
    int co = (int)(i & 63);
    long pix = i >> 6;
    int b = (int)(pix / (224 * 224));
    int r = (int)(pix - (long)b * 224 * 224);
    int h = r / 224, ww = r - h * 224;
    float acc = bias[co];
    for (int ci = 0; ci < 3; ++ci)
        for (int ky = 0; ky < 3; ++ky)
            for (int kx = 0; kx < 3; ++kx) {
                int hh = h + ky - 1, wx = ww + kx - 1;
                if (hh < 0 || hh >= 224 || wx < 0 || wx >= 224) continue;
                acc += x[(((size_t)b * 3 + ci) * 224 + hh) * 224 + wx] *
                       w[(((size_t)co * 3 + ci) * 3 + ky) * 3 + kx];
            }
    out[i] = f2bf(fmaxf(acc, 0.f));
}

// ---------------------------------------------------------------------------
// Weight repack: OIHW f32 -> [tap][coutTile16][cinChunk32][512 halves] bf16
//   lane L<16 : halves = W[cout=nt*16+L][cin kc*32 + 0..7, 16..23]
//   lane L>=16: halves = W[cout=nt*16+L-16][cin kc*32 + 8..15, 24..31]
// ---------------------------------------------------------------------------
__global__ void repack_w_kernel(const float* __restrict__ w, u16* __restrict__ out,
                                int CIN, int COUT) {
    long i = (long)blockIdx.x * 256 + threadIdx.x;
    long total = (long)9 * CIN * COUT;
    if (i >= total) return;
    int within = (int)(i & 511);
    long blk = i >> 9;
    int nKC = CIN / 32, nNT = COUT / 16;
    int kc = (int)(blk % nKC); blk /= nKC;
    int nt = (int)(blk % nNT); blk /= nNT;
    int tap = (int)blk;
    int laneq = within >> 4;
    int e = within & 15;
    int co = nt * 16 + (laneq & 15);
    int cin = kc * 32 + (laneq >> 4) * 8 + (e & 7) + (e >> 3) * 16;
    int ky = tap / 3, kx = tap - (tap / 3) * 3;
    float v = w[(((size_t)co * CIN + cin) * 3 + ky) * 3 + kx];
    out[i] = f2bf(v);
}

// ---------------------------------------------------------------------------
// Conv3x3 implicit GEMM, v_wmma_f32_16x16x32_bf16, 2M x 2N register blocking.
// Wave computes a 32(pixel) x 32(cout) tile; K = CIN*9 as (tap, 32-cin) chunks.
// Out-of-bounds taps read a zeroed page (branchless, no exec manipulation).
// ---------------------------------------------------------------------------
template<int CIN, int COUT, bool RELU>
__global__ __launch_bounds__(256)
void conv3x3_wmma(const u16* __restrict__ in, const u16* __restrict__ wrep,
                  const float* __restrict__ bias, u16* __restrict__ out,
                  const u16* __restrict__ zpage, int Bn, int H, int W) {
    static_assert(CIN % 32 == 0 && COUT % 32 == 0, "tile shapes");
    const int lane = threadIdx.x & 31;
    const int wv   = threadIdx.x >> 5;
    const int nT16 = COUT / 16;
    const int nTN  = COUT / 32;
    const int total = Bn * H * W;
    const int nTM  = (total + 31) >> 5;
    long tile = (long)blockIdx.x * 8 + wv;
    if (tile >= (long)nTM * nTN) return;            // wave-uniform
    const int mt = (int)(tile / nTN);
    const int nt = (int)(tile % nTN);
    const int hiHalf = lane >> 4;                   // 0: K 0-7/16-23, 1: K 8-15/24-31
    const int pRow = lane & 15;                     // M row within 16-tile
    const int o1 = hiHalf * 8;

    const int pix0 = mt * 32 + pRow;
    const int pix1 = pix0 + 16;
    const bool pv0 = pix0 < total;
    const bool pv1 = pix1 < total;
    int img0 = 0, h0 = 0, w0 = 0, img1 = 0, h1 = 0, w1 = 0;
    if (pv0) { img0 = pix0 / (H * W); int r = pix0 - img0 * H * W; h0 = r / W; w0 = r - h0 * W; }
    if (pv1) { img1 = pix1 / (H * W); int r = pix1 - img1 * H * W; h1 = r / W; w1 = r - h1 * W; }

    union AB { v16bf v; float4 f[2]; };
    v8f acc00 = {}, acc01 = {}, acc10 = {}, acc11 = {};

    for (int tap = 0; tap < 9; ++tap) {
        const int dy = tap / 3 - 1, dx = tap - (tap / 3) * 3 - 1;
        const int ha = h0 + dy, wa = w0 + dx;
        const int hb = h1 + dy, wb = w1 + dx;
        const bool inb0 = pv0 && ha >= 0 && ha < H && wa >= 0 && wa < W;
        const bool inb1 = pv1 && hb >= 0 && hb < H && wb >= 0 && wb < W;
        // branchless: out-of-bounds lanes read the zero page
        const u16* a0 = inb0 ? (in + (((size_t)img0 * H + ha) * W + wa) * CIN) : zpage;
        const u16* a1 = inb1 ? (in + (((size_t)img1 * H + hb) * W + wb) * CIN) : zpage;
        const u16* wb0 = wrep + (((size_t)tap * nT16 + 2 * nt) * (CIN / 32)) * 512 + (size_t)lane * 16;
        const u16* wb1 = wb0 + (size_t)(CIN / 32) * 512;
        __builtin_prefetch(wb0, 0, 1);
#pragma unroll
        for (int kc = 0; kc < CIN / 32; ++kc) {
            AB A0, A1, B0, B1;
            A0.f[0] = *(const float4*)(a0 + kc * 32 + o1);
            A0.f[1] = *(const float4*)(a0 + kc * 32 + o1 + 16);
            A1.f[0] = *(const float4*)(a1 + kc * 32 + o1);
            A1.f[1] = *(const float4*)(a1 + kc * 32 + o1 + 16);
            B0.f[0] = *(const float4*)(wb0 + (size_t)kc * 512);
            B0.f[1] = *(const float4*)(wb0 + (size_t)kc * 512 + 8);
            B1.f[0] = *(const float4*)(wb1 + (size_t)kc * 512);
            B1.f[1] = *(const float4*)(wb1 + (size_t)kc * 512 + 8);
            acc00 = __builtin_amdgcn_wmma_f32_16x16x32_bf16(false, A0.v, false, B0.v, (short)0, acc00, false, false);
            acc01 = __builtin_amdgcn_wmma_f32_16x16x32_bf16(false, A0.v, false, B1.v, (short)0, acc01, false, false);
            acc10 = __builtin_amdgcn_wmma_f32_16x16x32_bf16(false, A1.v, false, B0.v, (short)0, acc10, false, false);
            acc11 = __builtin_amdgcn_wmma_f32_16x16x32_bf16(false, A1.v, false, B1.v, (short)0, acc11, false, false);
        }
    }

    // D layout: VGPR r -> M = hiHalf*8 + r, N = lane&15
    const int coA = nt * 32 + pRow;
    const int coB = coA + 16;
    const float bA = bias[coA];
    const float bB = bias[coB];
#pragma unroll
    for (int r2 = 0; r2 < 8; ++r2) {
        int m = hiHalf * 8 + r2;
        int pA = mt * 32 + m;
        int pB = pA + 16;
        if (pA < total) {
            float v0 = acc00[r2] + bA;
            float v1 = acc01[r2] + bB;
            if (RELU) { v0 = fmaxf(v0, 0.f); v1 = fmaxf(v1, 0.f); }
            out[(size_t)pA * COUT + coA] = f2bf(v0);
            out[(size_t)pA * COUT + coB] = f2bf(v1);
        }
        if (pB < total) {
            float v0 = acc10[r2] + bA;
            float v1 = acc11[r2] + bB;
            if (RELU) { v0 = fmaxf(v0, 0.f); v1 = fmaxf(v1, 0.f); }
            out[(size_t)pB * COUT + coA] = f2bf(v0);
            out[(size_t)pB * COUT + coB] = f2bf(v1);
        }
    }
}

// ---------------------------------------------------------------------------
// Maxpool 2x2 NHWC bf16, 8 channels (16B) per thread
// ---------------------------------------------------------------------------
__global__ void maxpool2_nhwc(const u16* __restrict__ in, u16* __restrict__ out,
                              int Bn, int H, int W, int C) {
    long i = (long)blockIdx.x * 256 + threadIdx.x;
    int Ho = H >> 1, Wo = W >> 1, C8 = C >> 3;
    long total = (long)Bn * Ho * Wo * C8;
    if (i >= total) return;
    int c8 = (int)(i % C8); long p = i / C8;
    int b = (int)(p / ((long)Ho * Wo));
    int r = (int)(p - (long)b * Ho * Wo);
    int ho = r / Wo, wo = r - ho * Wo;
    const u16* b00 = in + (((size_t)b * H + 2 * ho) * W + 2 * wo) * C + (size_t)c8 * 8;
    const u16* b01 = b00 + C;
    const u16* b10 = b00 + (size_t)W * C;
    const u16* b11 = b10 + C;
    union V8 { uint4 v; u16 s[8]; };
    V8 x00, x01, x10, x11, o;
    x00.v = *(const uint4*)b00;
    x01.v = *(const uint4*)b01;
    x10.v = *(const uint4*)b10;
    x11.v = *(const uint4*)b11;
#pragma unroll
    for (int e = 0; e < 8; ++e) {
        float m = fmaxf(fmaxf(bf2f(x00.s[e]), bf2f(x01.s[e])),
                        fmaxf(bf2f(x10.s[e]), bf2f(x11.s[e])));
        o.s[e] = f2bf(m);
    }
    *(uint4*)(out + (size_t)i * 8) = o.v;
}

// ---------------------------------------------------------------------------
// Avgpool 14->7 NHWC bf16 -> f32 [B,49,512]
// ---------------------------------------------------------------------------
__global__ void avgpool2_to_f32(const u16* __restrict__ in, float* __restrict__ out, int Bn) {
    long i = (long)blockIdx.x * 256 + threadIdx.x;
    long total = (long)Bn * 49 * 512;
    if (i >= total) return;
    int c = (int)(i % 512); long p = i / 512;
    int b = (int)(p / 49); int r = (int)(p - (long)b * 49);
    int ph = r / 7, pw = r - ph * 7;
    float s = 0.f;
    for (int dy = 0; dy < 2; ++dy)
        for (int dx = 0; dx < 2; ++dx)
            s += bf2f(in[(((size_t)b * 14 + 2 * ph + dy) * 14 + 2 * pw + dx) * 512 + c]);
    out[i] = 0.25f * s;
}

// ---------------------------------------------------------------------------
// 1x1 conv "shr": fea[b,cc,d] = shr_b[cc] + sum_c avg[b,d,c] * shr_w[cc,c]
// ---------------------------------------------------------------------------
__global__ void shr_conv_kernel(const float* __restrict__ avg, const float* __restrict__ w,
                                const float* __restrict__ bias, float* __restrict__ fea, int Bn) {
    long i = (long)blockIdx.x * 256 + threadIdx.x;
    long total = (long)Bn * 128 * 49;
    if (i >= total) return;
    int d = (int)(i % 49); long r = i / 49;
    int cc = (int)(r % 128); int b = (int)(r / 128);
    const float* av = avg + ((size_t)b * 49 + d) * 512;
    const float* wr = w + (size_t)cc * 512;
    float s = bias[cc];
    for (int c = 0; c < 512; ++c) s += av[c] * wr[c];
    fea[i] = s;
}

// ---------------------------------------------------------------------------
// feedlayer: LN -> Linear -> GELU -> Linear on (fea + pos), also kn = l2(out)
// grid = B*128 rows, 64 threads
// ---------------------------------------------------------------------------
__global__ void feed_kernel(const float* __restrict__ fea, const float* __restrict__ pos,
                            const float* __restrict__ g, const float* __restrict__ be,
                            const float* __restrict__ fw1, const float* __restrict__ fb1,
                            const float* __restrict__ fw2, const float* __restrict__ fb2,
                            float* __restrict__ fap, float* __restrict__ kn) {
    __shared__ float z[49], zn[49], h1[49], red[64];
    int r = blockIdx.x; int c = r & 127; int t = threadIdx.x;
    if (t < 49) z[t] = fea[(size_t)r * 49 + t] + pos[(size_t)c * 49 + t];
    __syncthreads();
    float mean = block_reduce_sum<64>((t < 49) ? z[t] : 0.f, red) / 49.f;
    float dd = (t < 49) ? (z[t] - mean) : 0.f;
    float var = block_reduce_sum<64>(dd * dd, red) / 49.f;
    float rstd = rsqrtf(var + 1e-5f);
    if (t < 49) zn[t] = (z[t] - mean) * rstd * g[t] + be[t];
    __syncthreads();
    if (t < 49) {
        float a = fb1[t];
        for (int k = 0; k < 49; ++k) a += zn[k] * fw1[k * 49 + t];
        h1[t] = 0.5f * a * (1.f + erff(a * 0.70710678118654752f));
    }
    __syncthreads();
    if (t < 49) {
        float a = fb2[t];
        for (int k = 0; k < 49; ++k) a += h1[k] * fw2[k * 49 + t];
        fap[(size_t)r * 49 + t] = a;
        z[t] = a;
    }
    __syncthreads();
    float q = (t < 49) ? z[t] * z[t] : 0.f;
    float nrm = fmaxf(sqrtf(block_reduce_sum<64>(q, red)), 1e-12f);
    if (t < 49) kn[(size_t)r * 49 + t] = z[t] / nrm;
}

// ---------------------------------------------------------------------------
// CE attention: grid = B*20, 128 threads
// ---------------------------------------------------------------------------
__global__ void cls_attn_kernel(const float* __restrict__ cls, const float* __restrict__ kn,
                                const float* __restrict__ fap, const int* __restrict__ label,
                                float* __restrict__ o_cls_alpha, float* __restrict__ o_cls_pred,
                                float* __restrict__ o_max_v) {
    __shared__ float cn[49], al[128], red[128];
    int blk = blockIdx.x; int b = blk / 20; int q = blk % 20; int t = threadIdx.x;
    float s = (t < 49) ? cls[q * 49 + t] * cls[q * 49 + t] : 0.f;
    float inv = 1.f / fmaxf(sqrtf(block_reduce_sum<128>(s, red)), 1e-12f);
    if (t < 49) cn[t] = cls[q * 49 + t] * inv;
    __syncthreads();
    {
        float a = 0.f;
        const float* kr = kn + ((size_t)b * 128 + t) * 49;
        for (int d = 0; d < 49; ++d) a += cn[d] * kr[d];
        al[t] = a;
        o_cls_alpha[((size_t)b * 20 + q) * 128 + t] = a;
    }
    __syncthreads();
    float v = 0.f;
    if (t < 49) {
        for (int c = 0; c < 128; ++c) v += al[c] * fap[((size_t)b * 128 + c) * 49 + t];
    }
    float tot = block_reduce_sum<128>((t < 49) ? v : 0.f, red);
    if (t == 0) o_cls_pred[b * 20 + q] = tot / 49.f;
    if (t < 49 && q == label[b]) o_max_v[b * 49 + t] = v;
}

// ---------------------------------------------------------------------------
// Query generation MLP (cls_token -> all_q [20,245]); grid 20, 256 threads
// ---------------------------------------------------------------------------
__global__ void qgen_kernel(const float* __restrict__ cls,
                            const float* __restrict__ w1, const float* __restrict__ b1,
                            const float* __restrict__ w2, const float* __restrict__ b2,
                            const float* __restrict__ w3, const float* __restrict__ b3,
                            float* __restrict__ all_q) {
    __shared__ float in[49], h1s[256], h2s[256];
    int q = blockIdx.x, t = threadIdx.x;
    if (t < 49) in[t] = cls[q * 49 + t];
    __syncthreads();
    float a = b1[t];
    for (int k = 0; k < 49; ++k) a += in[k] * w1[k * 256 + t];
    h1s[t] = fmaxf(a, 0.f);
    __syncthreads();
    a = b2[t];
    for (int k = 0; k < 256; ++k) a += h1s[k] * w2[k * 256 + t];
    h2s[t] = fmaxf(a, 0.f);
    __syncthreads();
    if (t < 245) {
        a = b3[t];
        for (int k = 0; k < 256; ++k) a += h2s[k] * w3[k * 245 + t];
        all_q[q * 245 + t] = a;
    }
}

// ---------------------------------------------------------------------------
// Pair discriminator; grid 400 (=C*C), 256 threads
// ---------------------------------------------------------------------------
__global__ void pair_kernel(const float* __restrict__ cls, const float* __restrict__ all_q,
                            const float* __restrict__ w1, const float* __restrict__ b1,
                            const float* __restrict__ w2, const float* __restrict__ b2,
                            const float* __restrict__ w3, const float* __restrict__ b3,
                            float* __restrict__ pp) {
    __shared__ float in[294], h1s[256], h2s[256];
    int blk = blockIdx.x; int i = blk / 20, j = blk % 20; int t = threadIdx.x;
    if (t < 49) in[t] = cls[i * 49 + t];
    if (t < 245) in[49 + t] = all_q[j * 245 + t];
    __syncthreads();
    float a = b1[t];
    for (int k = 0; k < 294; ++k) a += in[k] * w1[k * 256 + t];
    h1s[t] = fmaxf(a, 0.f);
    __syncthreads();
    a = b2[t];
    for (int k = 0; k < 256; ++k) a += h1s[k] * w2[k * 256 + t];
    h2s[t] = fmaxf(a, 0.f);
    __syncthreads();
    if (t == 0) {
        float s = b3[0];
        for (int k = 0; k < 256; ++k) s += h2s[k] * w3[k];
        pp[blk] = s;
    }
}

__global__ void fqp_gather_kernel(const float* __restrict__ pp, const int* __restrict__ label,
                                  float* __restrict__ o_fqp, int Bn) {
    int i = blockIdx.x * 128 + threadIdx.x;
    if (i >= Bn * 20) return;
    int b = i / 20, q = i % 20;
    o_fqp[i] = pp[label[b] * 20 + q];
}

// ---------------------------------------------------------------------------
// Pattern attention; grid = B*5, 128 threads
// ---------------------------------------------------------------------------
__global__ void pat_attn_kernel(const float* __restrict__ all_q, const int* __restrict__ label,
                                const float* __restrict__ kn, const float* __restrict__ fap,
                                float* __restrict__ o_alpha, float* __restrict__ o_final_v,
                                float* __restrict__ normbuf) {
    __shared__ float qn[49], al[128], red[128];
    int blk = blockIdx.x; int b = blk / 5, p = blk % 5; int t = threadIdx.x;
    const float* qrow = all_q + (size_t)label[b] * 245 + p * 49;
    float s = (t < 49) ? qrow[t] * qrow[t] : 0.f;
    float inv = 1.f / fmaxf(sqrtf(block_reduce_sum<128>(s, red)), 1e-12f);
    if (t < 49) qn[t] = qrow[t] * inv;
    __syncthreads();
    {
        float a = 0.f;
        const float* kr = kn + ((size_t)b * 128 + t) * 49;
        for (int d = 0; d < 49; ++d) a += qn[d] * kr[d];
        al[t] = a;
        o_alpha[((size_t)b * 5 + p) * 128 + t] = a;
    }
    __syncthreads();
    float v = 0.f;
    if (t < 49) {
        for (int c = 0; c < 128; ++c) v += al[c] * fap[((size_t)b * 128 + c) * 49 + t];
        o_final_v[((size_t)b * 5 + p) * 49 + t] = v;
    }
    float tot = block_reduce_sum<128>((t < 49) ? v : 0.f, red);
    if (t == 0) normbuf[blk] = tot / 49.f;
}

// ---------------------------------------------------------------------------
// choose_value 'hard'; grid B, 64 threads
// ---------------------------------------------------------------------------
__global__ void choose_kernel(const float* __restrict__ normbuf, const float* __restrict__ final_v,
                              const float* __restrict__ all_q, const int* __restrict__ label,
                              float* __restrict__ o_choose, float* __restrict__ o_negq) {
    __shared__ int ind;
    int b = blockIdx.x; int t = threadIdx.x;
    if (t == 0) {
        float best = normbuf[b * 5]; int bi = 0;
        for (int p = 1; p < 5; ++p) {
            float v = normbuf[b * 5 + p];
            if (v > best) { best = v; bi = p; }
        }
        ind = bi;
    }
    __syncthreads();
    if (t < 49) o_choose[b * 49 + t] = final_v[((size_t)b * 5 + ind) * 49 + t];
    int o = 0;
    for (int p = 0; p < 5; ++p) {
        if (p == ind) continue;
        if (t < 49) o_negq[((size_t)b * 4 + o) * 49 + t] = all_q[(size_t)label[b] * 245 + p * 49 + t];
        ++o;
    }
}

// ---------------------------------------------------------------------------
// Host-side conv dispatch
// ---------------------------------------------------------------------------
template<int CI, int CO>
static void conv_layer(const u16* in, const u16* wrep, const float* bias, u16* out,
                       const u16* zpage, int Bn, int H, int W, bool relu, hipStream_t s) {
    int total = Bn * H * W;
    int mt = (total + 31) / 32;
    int nt = CO / 32;
    long tiles = (long)mt * nt;
    int blocks = (int)((tiles + 7) / 8);
    if (relu) conv3x3_wmma<CI, CO, true><<<blocks, 256, 0, s>>>(in, wrep, bias, out, zpage, Bn, H, W);
    else      conv3x3_wmma<CI, CO, false><<<blocks, 256, 0, s>>>(in, wrep, bias, out, zpage, Bn, H, W);
}

static void run_layer(int l, const u16* in, const u16* wrep, const float* bias, u16* out,
                      const u16* zpage, int Bn, int H, int W, hipStream_t s) {
    bool relu = (l < 15);
    int ci = CFG_CI[l], co = CFG_CO[l];
    if      (ci ==  64 && co ==  64) conv_layer< 64,  64>(in, wrep, bias, out, zpage, Bn, H, W, relu, s);
    else if (ci ==  64 && co == 128) conv_layer< 64, 128>(in, wrep, bias, out, zpage, Bn, H, W, relu, s);
    else if (ci == 128 && co == 128) conv_layer<128, 128>(in, wrep, bias, out, zpage, Bn, H, W, relu, s);
    else if (ci == 128 && co == 256) conv_layer<128, 256>(in, wrep, bias, out, zpage, Bn, H, W, relu, s);
    else if (ci == 256 && co == 256) conv_layer<256, 256>(in, wrep, bias, out, zpage, Bn, H, W, relu, s);
    else if (ci == 256 && co == 512) conv_layer<256, 512>(in, wrep, bias, out, zpage, Bn, H, W, relu, s);
    else if (ci == 512 && co == 512) conv_layer<512, 512>(in, wrep, bias, out, zpage, Bn, H, W, relu, s);
}

// ---------------------------------------------------------------------------
// kernel_launch
// ---------------------------------------------------------------------------
extern "C" void kernel_launch(void* const* d_in, const int* in_sizes, int n_in,
                              void* d_out, int out_size, void* d_ws, size_t ws_size,
                              hipStream_t stream) {
    (void)n_in; (void)out_size; (void)ws_size;
    const int Bn = in_sizes[0] / (3 * 224 * 224);     // 4

    // ---- inputs ----
    const float* x = (const float*)d_in[0];
    const float* vgg_w[16]; const float* vgg_b[16];
    for (int l = 0; l < 16; ++l) { vgg_w[l] = (const float*)d_in[1 + l]; vgg_b[l] = (const float*)d_in[17 + l]; }
    const float* shr_w   = (const float*)d_in[33];
    const float* shr_b   = (const float*)d_in[34];
    const float* pos_emb = (const float*)d_in[35];
    const float* ln_g    = (const float*)d_in[36];
    const float* ln_b    = (const float*)d_in[37];
    const float* fw1 = (const float*)d_in[38]; const float* fb1 = (const float*)d_in[39];
    const float* fw2 = (const float*)d_in[40]; const float* fb2 = (const float*)d_in[41];
    const float* qg_w1 = (const float*)d_in[42]; const float* qg_b1 = (const float*)d_in[43];
    const float* qg_w2 = (const float*)d_in[44]; const float* qg_b2 = (const float*)d_in[45];
    const float* qg_w3 = (const float*)d_in[46]; const float* qg_b3 = (const float*)d_in[47];
    const float* qd_w1 = (const float*)d_in[48]; const float* qd_b1 = (const float*)d_in[49];
    const float* qd_w2 = (const float*)d_in[50]; const float* qd_b2 = (const float*)d_in[51];
    const float* qd_w3 = (const float*)d_in[52]; const float* qd_b3 = (const float*)d_in[53];
    const float* cls_token = (const float*)d_in[54];
    const int*   label = (const int*)d_in[55];

    // ---- workspace layout ----
    char* ws = (char*)d_ws;
    auto alloc = [&](size_t bytes) -> void* {
        void* p = (void*)ws;
        ws += (bytes + 255) & ~(size_t)255;
        return p;
    };
    const size_t actBytes = (size_t)Bn * 224 * 224 * 64 * sizeof(u16);   // largest activation
    u16* bufA = (u16*)alloc(actBytes);
    u16* bufB = (u16*)alloc(actBytes);
    u16* zpage = (u16*)alloc(2048 * sizeof(u16));    // zero page (covers CIN<=1024 halves + slack)
    u16* wrepL[16] = {nullptr};
    for (int l = 1; l < 16; ++l)
        wrepL[l] = (u16*)alloc((size_t)9 * CFG_CI[l] * CFG_CO[l] * sizeof(u16));
    float* avgbuf  = (float*)alloc((size_t)Bn * 49 * 512 * sizeof(float));
    float* fea     = (float*)alloc((size_t)Bn * 128 * 49 * sizeof(float));
    float* fap     = (float*)alloc((size_t)Bn * 128 * 49 * sizeof(float));
    float* kn      = (float*)alloc((size_t)Bn * 128 * 49 * sizeof(float));
    float* all_q   = (float*)alloc((size_t)20 * 245 * sizeof(float));
    float* pp      = (float*)alloc((size_t)400 * sizeof(float));
    float* normbuf = (float*)alloc((size_t)Bn * 5 * sizeof(float));

    // ---- output sections (return-order flat) ----
    float* o = (float*)d_out;
    float* o_cls_pred  = o;               // [B,20]   80
    float* o_fqp       = o + 80;          // [B,20]   80
    float* o_choose    = o + 160;         // [B,49]   196
    float* o_final_v   = o + 356;         // [B,5,49] 980
    float* o_negq      = o + 1336;        // [B,4,49] 784
    float* o_cls_alpha = o + 2120;        // [B,20,128] 10240
    float* o_alpha     = o + 12360;       // [B,5,128]  2560
    float* o_max_v     = o + 14920;       // [B,49]   196

    // ---- 0) zero page init ----
    zero_u32_kernel<<<4, 256, 0, stream>>>((unsigned*)zpage, 1024);

    // ---- 1) weight repack for WMMA layers ----
    for (int l = 1; l < 16; ++l) {
        long n = (long)9 * CFG_CI[l] * CFG_CO[l];
        int blocks = (int)((n + 255) / 256);
        repack_w_kernel<<<blocks, 256, 0, stream>>>(vgg_w[l], wrepL[l], CFG_CI[l], CFG_CO[l]);
    }

    // ---- 2) layer 0 direct conv ----
    {
        long n = (long)Bn * 224 * 224 * 64;
        conv0_kernel<<<(int)((n + 255) / 256), 256, 0, stream>>>(x, vgg_w[0], vgg_b[0], bufA, Bn);
    }

    // ---- 3) WMMA conv chain with pools ----
    int H = 224, W = 224;
    u16* cur = bufA; u16* nxt = bufB;
    for (int l = 1; l < 16; ++l) {
        run_layer(l, cur, wrepL[l], vgg_b[l], nxt, zpage, Bn, H, W, stream);
        { u16* t = cur; cur = nxt; nxt = t; }
        if (l == 1 || l == 3 || l == 7 || l == 11) {
            int C = CFG_CO[l];
            long n = (long)Bn * (H / 2) * (W / 2) * (C >> 3);
            maxpool2_nhwc<<<(int)((n + 255) / 256), 256, 0, stream>>>(cur, nxt, Bn, H, W, C);
            H >>= 1; W >>= 1;
            { u16* t = cur; cur = nxt; nxt = t; }
        }
    }
    // cur: [Bn,14,14,512] bf16, final conv (no ReLU)

    // ---- 4) avgpool 14->7, shr 1x1 conv ----
    {
        long n = (long)Bn * 49 * 512;
        avgpool2_to_f32<<<(int)((n + 255) / 256), 256, 0, stream>>>(cur, avgbuf, Bn);
    }
    {
        long n = (long)Bn * 128 * 49;
        shr_conv_kernel<<<(int)((n + 255) / 256), 256, 0, stream>>>(avgbuf, shr_w, shr_b, fea, Bn);
    }

    // ---- 5) heads ----
    feed_kernel<<<Bn * 128, 64, 0, stream>>>(fea, pos_emb, ln_g, ln_b, fw1, fb1, fw2, fb2, fap, kn);
    cls_attn_kernel<<<Bn * 20, 128, 0, stream>>>(cls_token, kn, fap, label,
                                                 o_cls_alpha, o_cls_pred, o_max_v);
    qgen_kernel<<<20, 256, 0, stream>>>(cls_token, qg_w1, qg_b1, qg_w2, qg_b2, qg_w3, qg_b3, all_q);
    pair_kernel<<<400, 256, 0, stream>>>(cls_token, all_q, qd_w1, qd_b1, qd_w2, qd_b2, qd_w3, qd_b3, pp);
    fqp_gather_kernel<<<(Bn * 20 + 127) / 128, 128, 0, stream>>>(pp, label, o_fqp, Bn);
    pat_attn_kernel<<<Bn * 5, 128, 0, stream>>>(all_q, label, kn, fap, o_alpha, o_final_v, normbuf);
    choose_kernel<<<Bn, 64, 0, stream>>>(normbuf, o_final_v, all_q, label, o_choose, o_negq);
}
